// PixelContrastiveLearning_53704271069798
// MI455X (gfx1250) — compile-verified, matrix operand order checked by reading, not run on hardware
//
#include <hip/hip_runtime.h>
#include <hip/hip_bf16.h>
#include <math.h>

typedef __attribute__((ext_vector_type(16))) _Float16 v16h;
typedef __attribute__((ext_vector_type(8)))  float    v8f;

#define NB   4
#define CDIM 256
#define HW   25600
#define SS   1024
#define TEMP_INV 10.0f
#define TBT  1.4285714285714286f   /* TEMP / BASE_TEMP */
#define SIMTHR 0.7f

// ---------------------------------------------------------------------------
// WMMA helpers (gfx1250, wave32). f16 inputs, f32 accumulate.
// ---------------------------------------------------------------------------
__device__ __forceinline__ v8f wmma_f16(v16h a, v16h b, v8f c) {
  return __builtin_amdgcn_wmma_f32_16x16x32_f16(
      /*neg_a=*/false, a, /*neg_b=*/false, b,
      /*c_mod=*/(short)0, c, /*reuse_a=*/false, /*reuse_b=*/false);
}

// Load a 16x32 f16 fragment from a row-major matrix (ld elements per row).
// ISA 7.12.2 16-bit A layout: lanes 0-15 hold rows (M or N) with K 0-7 &
// 16-23; lanes 16-31 hold K 8-15 & 24-31. B fragment uses the same mapping
// with N as the "row", which matches (x @ W.T): both A and B read contiguous
// K-runs from row-major [row][k] storage.
__device__ __forceinline__ v16h frag_ld(const _Float16* base, int ld, int row, int koff) {
  int lane = threadIdx.x & 31;
  const _Float16* p = base + (size_t)(row + (lane & 15)) * ld + koff + ((lane >> 4) << 3);
  union { v16h v; _Float16 e[16]; } u;
#pragma unroll
  for (int i = 0; i < 8; ++i) {
    int k = ((i & 4) ? 16 : 0) + ((i & 3) << 1);
    u.e[2 * i]     = p[k];
    u.e[2 * i + 1] = p[k + 1];
  }
  return u.v;
}

// ---------------------------------------------------------------------------
// K0: cast weights f32->f16, zero per-batch accumulators
// ---------------------------------------------------------------------------
__global__ void k_prep(const float* w1, const float* w2, const float* wa1,
                       _Float16* w1h, _Float16* w2h, _Float16* wa1h, float* acc) {
  int tid = blockIdx.x * blockDim.x + threadIdx.x;
  int stride = gridDim.x * blockDim.x;
  for (int i = tid; i < CDIM * CDIM; i += stride) {
    w1h[i] = (_Float16)w1[i];
    w2h[i] = (_Float16)w2[i];
  }
  for (int i = tid; i < 64 * CDIM; i += stride) wa1h[i] = (_Float16)wa1[i];
  if (tid < 2 * NB) acc[tid] = 0.f;
}

// ---------------------------------------------------------------------------
// K1: stable partition of labels -> idx[1024], sel[1024] per batch.
// Equivalent to argsort(where(valid,0,1), stable)[:1024].
// ---------------------------------------------------------------------------
__global__ __launch_bounds__(1024) void k_select(const int* labels, int* idx, int* sel) {
  __shared__ int scnt[1024];
  int b = blockIdx.x;
  int t = threadIdx.x;
  const int CH = HW / 1024; // 25
  const int* lab = labels + b * HW;
  int base = t * CH;
  int cnt = 0;
  for (int i = 0; i < CH; ++i) cnt += (lab[base + i] == 1);
  scnt[t] = cnt;
  __syncthreads();
  // inclusive Hillis-Steele scan
  for (int off = 1; off < 1024; off <<= 1) {
    int v = (t >= off) ? scnt[t - off] : 0;
    __syncthreads();
    scnt[t] += v;
    __syncthreads();
  }
  int total = scnt[1023];      // total valid in this batch
  int excl  = scnt[t] - cnt;   // valid strictly before my chunk
  int vrun = 0;
  for (int i = 0; i < CH; ++i) {
    int p = base + i;
    int isv = (lab[p] == 1);
    int vbefore = excl + vrun;
    if (isv) {
      if (vbefore < SS) { idx[b * SS + vbefore] = p; sel[b * SS + vbefore] = 1; }
      vrun++;
    } else {
      int irank = p - vbefore;      // rank among invalid positions
      int dst = total + irank;      // fillers go after all valid
      if (dst < SS) { idx[b * SS + dst] = p; sel[b * SS + dst] = 0; }
    }
  }
}

// ---------------------------------------------------------------------------
// K2: gather selected pixel rows from (B,C,H,W) feats into f16 [SS][CDIM],
//     plus their superpixel ids.
// ---------------------------------------------------------------------------
__global__ void k_gather(const float* feats, const int* spx, const int* idx,
                         _Float16* xh, int* tsp) {
  int tid = blockIdx.x * blockDim.x + threadIdx.x;
  int n = NB * SS * CDIM;
  int stride = gridDim.x * blockDim.x;
  for (; tid < n; tid += stride) {
    int c = tid & (CDIM - 1);
    int r = (tid >> 8) & (SS - 1);
    int b = tid >> 18;
    int p = idx[b * SS + r];
    xh[tid] = (_Float16)feats[((size_t)(b * CDIM + c)) * HW + p];
    if (c == 0) tsp[b * SS + r] = spx[b * HW + p];
  }
}

// ---------------------------------------------------------------------------
// K3: projection MLP + row L2-norm + attention head, all via WMMA.
// One workgroup (128 threads = 4 waves) per 64 selected rows.
// ---------------------------------------------------------------------------
__global__ __launch_bounds__(128) void k_proj(const _Float16* xh,
                                              const _Float16* w1h, const float* b1,
                                              const _Float16* w2h, const float* b2,
                                              const _Float16* wa1h, const float* ba1,
                                              const float* wa2, const float* ba2,
                                              _Float16* fnh, float* tw) {
  __shared__ _Float16 h1[64 * CDIM];   // ReLU(X W1^T + b1), f16
  __shared__ _Float16 p16[64 * CDIM];  // P = H1 W2^T + b2, f16 (pre-norm)
  __shared__ _Float16 a1[64 * 64];     // ReLU(X Wa1^T + ba1)
  int b    = blockIdx.x >> 4;
  int mblk = blockIdx.x & 15;
  int wave = threadIdx.x >> 5;
  int lane = threadIdx.x & 31;
  int lm = lane & 15, lh = lane >> 4;
  int m0 = mblk * 64 + wave * 16;              // row base within batch
  const _Float16* X = xh + (size_t)b * SS * CDIM;

  // Keep full-K A-fragments in registers; reused for GEMM1 and attention.
  v16h af[8];
#pragma unroll
  for (int kb = 0; kb < 8; ++kb) af[kb] = frag_ld(X, CDIM, m0, kb * 32);

  // GEMM1: H1 = relu(X @ W1^T + b1)
  for (int nt = 0; nt < 16; ++nt) {
    v8f c = {};
#pragma unroll
    for (int kb = 0; kb < 8; ++kb)
      c = wmma_f16(af[kb], frag_ld(w1h, CDIM, nt * 16, kb * 32), c);
#pragma unroll
    for (int r = 0; r < 8; ++r) {
      int M = wave * 16 + r + (lh << 3);
      int N = nt * 16 + lm;
      float v = c[r] + b1[N];
      h1[M * CDIM + N] = (_Float16)(v > 0.f ? v : 0.f);
    }
  }
  // Attention GEMM: A1 = relu(X @ Wa1^T + ba1)  (reuses af)
  for (int nt = 0; nt < 4; ++nt) {
    v8f c = {};
#pragma unroll
    for (int kb = 0; kb < 8; ++kb)
      c = wmma_f16(af[kb], frag_ld(wa1h, CDIM, nt * 16, kb * 32), c);
#pragma unroll
    for (int r = 0; r < 8; ++r) {
      int M = wave * 16 + r + (lh << 3);
      int N = nt * 16 + lm;
      float v = c[r] + ba1[N];
      a1[M * 64 + N] = (_Float16)(v > 0.f ? v : 0.f);
    }
  }
  __syncthreads();
  // GEMM2: P = H1 @ W2^T + b2
  v16h hf[8];
#pragma unroll
  for (int kb = 0; kb < 8; ++kb) hf[kb] = frag_ld(h1, CDIM, wave * 16, kb * 32);
  for (int nt = 0; nt < 16; ++nt) {
    v8f c = {};
#pragma unroll
    for (int kb = 0; kb < 8; ++kb)
      c = wmma_f16(hf[kb], frag_ld(w2h, CDIM, nt * 16, kb * 32), c);
#pragma unroll
    for (int r = 0; r < 8; ++r) {
      int M = wave * 16 + r + (lh << 3);
      int N = nt * 16 + lm;
      p16[M * CDIM + N] = (_Float16)(c[r] + b2[N]);
    }
  }
  __syncthreads();
  // Row L2-normalize + attention logit/sigmoid. One thread per row.
  if (threadIdx.x < 64) {
    int lrow = threadIdx.x;
    int row = mblk * 64 + lrow;
    float ssq = 0.f;
    for (int k = 0; k < CDIM; ++k) { float v = (float)p16[lrow * CDIM + k]; ssq += v * v; }
    float inv = 1.f / fmaxf(sqrtf(ssq), 1e-12f);
    _Float16* dst = fnh + ((size_t)b * SS + row) * CDIM;
    for (int k = 0; k < CDIM; ++k) dst[k] = (_Float16)((float)p16[lrow * CDIM + k] * inv);
    float al = ba2[0];
    for (int k = 0; k < 64; ++k) al += (float)a1[lrow * 64 + k] * wa2[k];
    tw[b * SS + row] = 1.f / (1.f + __expf(-al));
  }
}

// ---------------------------------------------------------------------------
// K4: sim = FN @ FN^T per batch, 64x64 tile per workgroup via WMMA.
// ---------------------------------------------------------------------------
__global__ __launch_bounds__(128) void k_sim(const _Float16* fnh, float* sim) {
  int wg = blockIdx.x;
  int b  = wg >> 8;
  int mb = (wg >> 4) & 15;
  int nb = wg & 15;
  int wave = threadIdx.x >> 5, lane = threadIdx.x & 31;
  int lm = lane & 15, lh = lane >> 4;
  const _Float16* F = fnh + (size_t)b * SS * CDIM;
  int m0 = mb * 64 + wave * 16;
  v16h af[8];
#pragma unroll
  for (int kb = 0; kb < 8; ++kb) af[kb] = frag_ld(F, CDIM, m0, kb * 32);
  float* S = sim + (size_t)b * SS * SS;
  for (int nt = 0; nt < 4; ++nt) {
    int n0 = nb * 64 + nt * 16;
    v8f c = {};
#pragma unroll
    for (int kb = 0; kb < 8; ++kb)
      c = wmma_f16(af[kb], frag_ld(F, CDIM, n0, kb * 32), c);
#pragma unroll
    for (int r = 0; r < 8; ++r) {
      int M = m0 + r + (lh << 3);
      int N = n0 + lm;
      S[(size_t)M * SS + N] = c[r];
    }
  }
}

// ---------------------------------------------------------------------------
// K5: per-row denom + masked weighted log-prob sums -> per-batch accumulators.
// ---------------------------------------------------------------------------
__global__ __launch_bounds__(256) void k_loss(const float* sim, const int* sel,
                                              const int* tsp, const float* tw,
                                              float* acc) {
  __shared__ float red[256];
  int b = blockIdx.x >> 10;
  int i = blockIdx.x & (SS - 1);
  const float* Si  = sim + ((size_t)b * SS + i) * SS;
  const int*   sb  = sel + b * SS;
  const int*   tb  = tsp + b * SS;
  const float* wb  = tw  + b * SS;
  int t = threadIdx.x;

  float d = 0.f;
  for (int j = t; j < SS; j += 256) d += sb[j] ? __expf(Si[j] * TEMP_INV) : 0.f;
  red[t] = d; __syncthreads();
  for (int off = 128; off > 0; off >>= 1) { if (t < off) red[t] += red[t + off]; __syncthreads(); }
  float denom = red[0];
  __syncthreads();
  float lg = logf(denom > 0.f ? denom : 1.f);

  float num = 0.f, ps = 0.f;
  if (sb[i]) {
    int   ti = tb[i];
    float wi = wb[i];
    for (int j = t; j < SS; j += 256) {
      float s = Si[j];
      if (j != i && sb[j] && tb[j] == ti && s > SIMTHR) {
        float pw = wi * wb[j];
        num += pw * (s * TEMP_INV - lg);
        ps  += pw;
      }
    }
  }
  red[t] = num; __syncthreads();
  for (int off = 128; off > 0; off >>= 1) { if (t < off) red[t] += red[t + off]; __syncthreads(); }
  float tnum = red[0]; __syncthreads();
  red[t] = ps; __syncthreads();
  for (int off = 128; off > 0; off >>= 1) { if (t < off) red[t] += red[t + off]; __syncthreads(); }
  float tps = red[0];
  if (t == 0 && (tnum != 0.f || tps != 0.f)) {
    atomicAdd(&acc[b * 2 + 0], tnum);
    atomicAdd(&acc[b * 2 + 1], tps);
  }
}

// ---------------------------------------------------------------------------
// K6: combine per-batch losses into the scalar output.
// ---------------------------------------------------------------------------
__global__ void k_final(const float* acc, float* out) {
  if (threadIdx.x == 0 && blockIdx.x == 0) {
    float s = 0.f, n = 0.f;
    for (int b = 0; b < NB; ++b) {
      float ps = acc[b * 2 + 1];
      if (ps > 0.f) { s += -TBT * acc[b * 2 + 0] / ps; n += 1.f; }
    }
    out[0] = (n > 0.f) ? s / fmaxf(n, 1.f) : 0.f;
  }
}

// ---------------------------------------------------------------------------
extern "C" void kernel_launch(void* const* d_in, const int* in_sizes, int n_in,
                              void* d_out, int out_size, void* d_ws, size_t ws_size,
                              hipStream_t stream) {
  (void)in_sizes; (void)n_in; (void)out_size; (void)ws_size;
  const float* feats = (const float*)d_in[0];
  const int*   labels= (const int*)d_in[1];
  const int*   spx   = (const int*)d_in[2];
  const float* w1    = (const float*)d_in[3];
  const float* b1    = (const float*)d_in[4];
  const float* w2    = (const float*)d_in[5];
  const float* b2    = (const float*)d_in[6];
  const float* wa1   = (const float*)d_in[7];
  const float* ba1   = (const float*)d_in[8];
  const float* wa2   = (const float*)d_in[9];
  const float* ba2   = (const float*)d_in[10];
  float* out = (float*)d_out;

  char* ws = (char*)d_ws;
  size_t off = 0;
  auto carve = [&](size_t bytes) {
    char* p = ws + off;
    off += (bytes + 255) & ~(size_t)255;
    return p;
  };
  int*      idx  = (int*)     carve((size_t)NB * SS * 4);
  int*      sel  = (int*)     carve((size_t)NB * SS * 4);
  int*      tsp  = (int*)     carve((size_t)NB * SS * 4);
  float*    tw   = (float*)   carve((size_t)NB * SS * 4);
  float*    acc  = (float*)   carve((size_t)2 * NB * 4);
  _Float16* w1h  = (_Float16*)carve((size_t)CDIM * CDIM * 2);
  _Float16* w2h  = (_Float16*)carve((size_t)CDIM * CDIM * 2);
  _Float16* wa1h = (_Float16*)carve((size_t)64 * CDIM * 2);
  _Float16* xh   = (_Float16*)carve((size_t)NB * SS * CDIM * 2);
  _Float16* fnh  = (_Float16*)carve((size_t)NB * SS * CDIM * 2);
  float*    sim  = (float*)   carve((size_t)NB * SS * SS * 4);

  k_prep  <<<64, 256, 0, stream>>>(w1, w2, wa1, w1h, w2h, wa1h, acc);
  k_select<<<NB, 1024, 0, stream>>>(labels, idx, sel);
  k_gather<<<2048, 256, 0, stream>>>(feats, spx, idx, xh, tsp);
  k_proj  <<<NB * 16, 128, 0, stream>>>(xh, w1h, b1, w2h, b2, wa1h, ba1, wa2, ba2, fnh, tw);
  k_sim   <<<NB * 256, 128, 0, stream>>>(fnh, sim);
  k_loss  <<<NB * SS, 256, 0, stream>>>(sim, sel, tsp, tw, acc);
  k_final <<<1, 32, 0, stream>>>(acc, out);
}
